// HierarchicalPooling_10462540333072
// MI455X (gfx1250) — matrix operand order repaired and checked by reading, not compiled, not run on hardware
//
#include <hip/hip_runtime.h>
#include <hip/hip_bf16.h>

// CDNA5 (gfx1250) fused hierarchical-pooling kernel.
// bf16 WMMA (f32 accumulate) for all three GEMMs, single pass over x.

typedef __attribute__((ext_vector_type(16))) __bf16        v16bf;
typedef __attribute__((ext_vector_type(8)))  float         v8f;
typedef __attribute__((ext_vector_type(4)))  unsigned int  v4u;

union FragB16 {
    v16bf v;
    v4u   q[2];
};

__device__ __forceinline__ unsigned short f2bf(float f) {
    // hardware f32 -> bf16 convert (RTNE), stored as raw bits
    __bf16 h = (__bf16)f;
    return __builtin_bit_cast(unsigned short, h);
}

#define NC   128   // in_channels
#define NK   32    // super nodes
#define NB   64    // graphs
#define TILE 32    // nodes per block

// ---------------------------------------------------------------------------
// Prep: zero the atomic-accumulated `out` region and build bf16 transposed
// weights in workspace:  w1t[out=128][in=128], w2t[out=32][in=128].
// Transposed layout makes each WMMA B-fragment two contiguous 16B loads.
// ---------------------------------------------------------------------------
__global__ void hp_prep(const float* __restrict__ W1, const float* __restrict__ W2,
                        unsigned short* __restrict__ w1t, unsigned short* __restrict__ w2t,
                        float* __restrict__ out) {
    int tid    = blockIdx.x * blockDim.x + threadIdx.x;
    int stride = gridDim.x * blockDim.x;
    for (int i = tid; i < NB * NK * NC; i += stride) out[i] = 0.f;
    for (int i = tid; i < NC * NC; i += stride) {
        int j = i >> 7, k = i & 127;                 // j: out col, k: in row
        w1t[j * NC + k] = f2bf(W1[k * NC + j]);
    }
    for (int i = tid; i < NK * NC; i += stride) {
        int j = i >> 7, k = i & 127;
        w2t[j * NC + k] = f2bf(W2[k * NK + j]);
    }
}

// ---------------------------------------------------------------------------
// Fused: per 32-node tile -> GEMM1(+bias,ReLU) -> GEMM2(+bias) -> softmax ->
// write s -> GEMM3 (s^T @ x) with atomic segment scatter into out[graph].
// ---------------------------------------------------------------------------
__global__ void __launch_bounds__(256) hp_fused(
        const float* __restrict__ x, const int* __restrict__ batch,
        const float* __restrict__ b1, const float* __restrict__ b2,
        const unsigned short* __restrict__ w1t, const unsigned short* __restrict__ w2t,
        float* __restrict__ out, float* __restrict__ s_out, int N) {

    __shared__ __align__(16) unsigned short xs[TILE][NC];   // x tile, bf16
    __shared__ __align__(16) unsigned short xt[NC][TILE];   // x tile transposed
    __shared__ __align__(16) unsigned short hs[TILE][NC];   // hidden, bf16
    __shared__ __align__(16) float          lg[TILE][NK];   // logits -> probs f32
    __shared__ __align__(16) unsigned short st[NK][TILE];   // s transposed, bf16
    __shared__ __align__(16) unsigned short stm[NK][TILE];  // masked s^T
    __shared__ int bb[TILE];

    const int tid   = threadIdx.x;
    const int lane  = tid & 31;
    const int wv    = tid >> 5;        // 8 waves
    const int half  = lane >> 4;       // hi/lo 16-lane group
    const int lx    = lane & 15;
    const int node0 = blockIdx.x * TILE;

    // ---- stage x tile (f32 -> bf16, row-major + transposed) ----
    for (int i = tid; i < TILE * NC; i += 256) {
        int r = i >> 7, c = i & 127;
        int node = node0 + r;
        float f = (node < N) ? x[node * NC + c] : 0.f;
        unsigned short bv = f2bf(f);
        xs[r][c] = bv;
        xt[c][r] = bv;
    }
    if (tid < TILE) {
        int node = node0 + tid;
        bb[tid] = (node < N) ? batch[node] : 0;
    }
    __syncthreads();

    // ---- GEMM1: hs = relu(xs @ W1 + b1). wave wv owns columns [16wv,16wv+16).
    // B fragment (W1^T slice) is shared by both M-halves: load once per kk.
    {
        const int n = wv * 16 + lx;
        const float bias = b1[n];
        v8f acc0, acc1;
        #pragma unroll
        for (int r = 0; r < 8; ++r) { acc0[r] = bias; acc1[r] = bias; }
        #pragma unroll
        for (int kk = 0; kk < 4; ++kk) {
            FragB16 b;
            const unsigned short* bp = &w1t[n * NC + kk * 32 + half * 8];
            b.q[0] = *reinterpret_cast<const v4u*>(bp);
            b.q[1] = *reinterpret_cast<const v4u*>(bp + 16);

            FragB16 a0, a1;
            const unsigned short* ap0 = &xs[lx][kk * 32 + half * 8];
            a0.q[0] = *reinterpret_cast<const v4u*>(ap0);
            a0.q[1] = *reinterpret_cast<const v4u*>(ap0 + 16);
            const unsigned short* ap1 = &xs[16 + lx][kk * 32 + half * 8];
            a1.q[0] = *reinterpret_cast<const v4u*>(ap1);
            a1.q[1] = *reinterpret_cast<const v4u*>(ap1 + 16);

            acc0 = __builtin_amdgcn_wmma_f32_16x16x32_bf16(
                    false, a0.v, false, b.v, (short)0, acc0, false, false);
            acc1 = __builtin_amdgcn_wmma_f32_16x16x32_bf16(
                    false, a1.v, false, b.v, (short)0, acc1, false, false);
        }
        #pragma unroll
        for (int r = 0; r < 8; ++r) {
            hs[r + 8 * half][n]      = f2bf(fmaxf(acc0[r], 0.f));   // ReLU
            hs[16 + r + 8 * half][n] = f2bf(fmaxf(acc1[r], 0.f));
        }
    }
    __syncthreads();

    // ---- GEMM2: logits = hs @ W2 + b2 (waves 0..3 cover 32x32 outputs) ----
    if (wv < 4) {
        const int mh = wv >> 1, nt = wv & 1;
        const int n = nt * 16 + lx;
        const float bias = b2[n];
        v8f acc;
        #pragma unroll
        for (int r = 0; r < 8; ++r) acc[r] = bias;
        #pragma unroll
        for (int kk = 0; kk < 4; ++kk) {
            FragB16 a, b;
            const unsigned short* ap = &hs[mh * 16 + lx][kk * 32 + half * 8];
            a.q[0] = *reinterpret_cast<const v4u*>(ap);
            a.q[1] = *reinterpret_cast<const v4u*>(ap + 16);
            const unsigned short* bp = &w2t[n * NC + kk * 32 + half * 8];
            b.q[0] = *reinterpret_cast<const v4u*>(bp);
            b.q[1] = *reinterpret_cast<const v4u*>(bp + 16);
            acc = __builtin_amdgcn_wmma_f32_16x16x32_bf16(
                    false, a.v, false, b.v, (short)0, acc, false, false);
        }
        #pragma unroll
        for (int r = 0; r < 8; ++r)
            lg[mh * 16 + r + 8 * half][n] = acc[r];
    }
    __syncthreads();

    // ---- softmax: wave 0, one row (node) per lane ----
    if (tid < TILE) {
        float m = -3.4e38f;
        #pragma unroll
        for (int c = 0; c < NK; ++c) m = fmaxf(m, lg[tid][c]);
        float sum = 0.f;
        #pragma unroll
        for (int c = 0; c < NK; ++c) sum += __expf(lg[tid][c] - m);
        float inv = 1.0f / sum;
        #pragma unroll
        for (int c = 0; c < NK; ++c) {
            float p = __expf(lg[tid][c] - m) * inv;
            lg[tid][c] = p;          // f32 copy for s output
            st[c][tid] = f2bf(p);    // transposed bf16 for GEMM3 A-fragment
        }
    }
    __syncthreads();

    // ---- write s (f32), coalesced ----
    for (int i = tid; i < TILE * NK; i += 256) {
        int r = i >> 5, c = i & 31;
        int node = node0 + r;
        if (node < N) s_out[node * NK + c] = lg[r][c];
    }

    // ---- GEMM3: out[g] += s^T @ x over this tile's nodes.
    // A = s^T [32 classes x 32 nodes], B = x tile [32 nodes x 128 ch].
    // B fragment depends only on the wave's channel tile: load once.
    const int g0 = bb[0], g1 = bb[TILE - 1];
    {
        const int n = wv * 16 + lx;   // channel column
        FragB16 b;
        const unsigned short* bp = &xt[n][half * 8];
        b.q[0] = *reinterpret_cast<const v4u*>(bp);
        b.q[1] = *reinterpret_cast<const v4u*>(bp + 16);

        if (g0 == g1) {
            // fast path: whole tile belongs to one graph (common: batch sorted)
            #pragma unroll
            for (int mh = 0; mh < 2; ++mh) {
                v8f acc;
                #pragma unroll
                for (int r = 0; r < 8; ++r) acc[r] = 0.f;
                FragB16 a;
                const unsigned short* ap = &st[mh * 16 + lx][half * 8];
                a.q[0] = *reinterpret_cast<const v4u*>(ap);
                a.q[1] = *reinterpret_cast<const v4u*>(ap + 16);
                acc = __builtin_amdgcn_wmma_f32_16x16x32_bf16(
                        false, a.v, false, b.v, (short)0, acc, false, false);
                #pragma unroll
                for (int r = 0; r < 8; ++r) {
                    int cls = mh * 16 + r + 8 * half;
                    atomicAdd(&out[(g0 * NK + cls) * NC + n], acc[r]);
                }
            }
        } else {
            // boundary tile: loop over the graphs it spans with masked s^T
            for (int g = g0; g <= g1; ++g) {
                __syncthreads();   // previous iter's reads done before overwrite
                for (int i = tid; i < NK * TILE; i += 256) {
                    int cls = i >> 5, nd = i & 31;
                    stm[cls][nd] = (bb[nd] == g) ? st[cls][nd] : (unsigned short)0;
                }
                __syncthreads();
                #pragma unroll
                for (int mh = 0; mh < 2; ++mh) {
                    v8f acc;
                    #pragma unroll
                    for (int r = 0; r < 8; ++r) acc[r] = 0.f;
                    FragB16 a;
                    const unsigned short* ap = &stm[mh * 16 + lx][half * 8];
                    a.q[0] = *reinterpret_cast<const v4u*>(ap);
                    a.q[1] = *reinterpret_cast<const v4u*>(ap + 16);
                    acc = __builtin_amdgcn_wmma_f32_16x16x32_bf16(
                            false, a.v, false, b.v, (short)0, acc, false, false);
                    #pragma unroll
                    for (int r = 0; r < 8; ++r) {
                        int cls = mh * 16 + r + 8 * half;
                        atomicAdd(&out[(g * NK + cls) * NC + n], acc[r]);
                    }
                }
            }
        }
    }
}

// ---------------------------------------------------------------------------
extern "C" void kernel_launch(void* const* d_in, const int* in_sizes, int n_in,
                              void* d_out, int out_size, void* d_ws, size_t ws_size,
                              hipStream_t stream) {
    const float* x     = (const float*)d_in[0];
    const int*   batch = (const int*)  d_in[1];
    const float* W1    = (const float*)d_in[2];
    const float* b1    = (const float*)d_in[3];
    const float* W2    = (const float*)d_in[4];
    const float* b2    = (const float*)d_in[5];

    float* out   = (float*)d_out;                 // [B, K, C]
    float* s_out = out + NB * NK * NC;            // [N, K]

    unsigned short* w1t = (unsigned short*)d_ws;  // [128][128] bf16
    unsigned short* w2t = w1t + NC * NC;          // [32][128]  bf16

    const int N = in_sizes[1];                    // batch array length = #nodes

    hp_prep<<<256, 256, 0, stream>>>(W1, W2, w1t, w2t, out);
    hp_fused<<<(N + TILE - 1) / TILE, 256, 0, stream>>>(
        x, batch, b1, b2, w1t, w2t, out, s_out, N);
}
